// CTViT_23776938951221
// MI455X (gfx1250) — compile-verified, matrix operand order checked by reading, not run on hardware
//
#include <hip/hip_runtime.h>
#include <hip/hip_bf16.h>

// ---------------------------------------------------------------------------
// Types for CDNA5 WMMA (wave32, 16x16x32 f16 -> f32)
// ---------------------------------------------------------------------------
typedef __attribute__((ext_vector_type(16))) _Float16 v16h;
typedef __attribute__((ext_vector_type(8)))  float    v8f;
typedef __attribute__((ext_vector_type(4)))  unsigned v4u;

union Frag {
    v16h v;
    v4u  q[2];
    unsigned u[8];
};

#define HEADS 8
#define DH    64
#define DMODEL 512
#define ROWS  17408           // b * t * h * w = 4*17*16*16   (= 136 * 128)
#define CHUNK 4352            // ROWS / 4 (FFN row chunk, = 34 * 128)
#define SCORE_CHUNK 2176      // ROWS / 8 (VQ score chunk, = 17 * 128)
#define FFH   2730
#define FFH_PAD 2816          // FFN hidden width padded to multiple of 128
#define FFI_  1365
#define FFI_PAD 1376          // gated width padded to multiple of 32
#define CBN   8192

// ---------------------------------------------------------------------------
// Guard-free tiled WMMA GEMM with async global->LDS double buffering.
// C[M,N] = A[M,K] @ BT[N,K]^T (+bias[n]) (+res[m,n]).
// Preconditions (arranged by host): M % 128 == 0, N % 128 == 0, K % 32 == 0,
// A/BT 16B aligned with pitches K. BM=BN=128, BK=32, 256 threads (8 waves);
// each wave computes 32x64 => 8 x v_wmma_f32_16x16x32_f16 per K-step.
// Tile staging uses GLOBAL_LOAD_ASYNC_TO_LDS_B128 (ASYNCcnt) into the ping
// buffer while WMMAs consume the pong buffer.
// ---------------------------------------------------------------------------
#define BM 128
#define BN 128
#define BK 32
#define LP 40   // LDS pitch in halves (80B: 16B-aligned rows, conflict-free)

__device__ __forceinline__ void async_ld128(unsigned lds_addr, const void* gaddr) {
    asm volatile("global_load_async_to_lds_b128 %0, %1, off"
                 :: "v"(lds_addr), "v"(gaddr) : "memory");
}
__device__ __forceinline__ void wait_async0() {
    asm volatile("s_wait_asynccnt 0x0" ::: "memory");
}

template <typename OUT_T>
__global__ __launch_bounds__(256) void gemm_kernel(
    const _Float16* __restrict__ A, const _Float16* __restrict__ BT,
    OUT_T* __restrict__ C, int M, int N, int K,
    const float* __restrict__ bias, const float* __restrict__ residual)
{
    __shared__ _Float16 As[2][BM][LP];   // [buf][m][k]
    __shared__ _Float16 Bs[2][BN][LP];   // [buf][n][k]

    const int tid   = threadIdx.x;
    const int lane  = tid & 31;
    const int wave  = tid >> 5;
    const int waveM = wave >> 1;     // 0..3 -> 32 rows each
    const int waveN = wave & 1;      // 0..1 -> 64 cols each
    const int m0 = blockIdx.y * BM;
    const int n0 = blockIdx.x * BN;
    const int hi = lane >> 4;        // lane half
    const int nl = lane & 15;

    v8f acc[2][4];
#pragma unroll
    for (int mi = 0; mi < 2; ++mi)
#pragma unroll
        for (int nj = 0; nj < 4; ++nj)
#pragma unroll
            for (int e = 0; e < 8; ++e) acc[mi][nj][e] = 0.0f;

    // staging coordinates: 128x32 halves = 512 b128 chunks, 2 per thread per matrix
    const int rr0 = tid >> 2;            // 0..63   (row, +64 for second chunk)
    const int c8  = (tid & 3) * 8;       // 0,8,16,24

    const _Float16* ga0 = &A [(size_t)(m0 + rr0     ) * K + c8];
    const _Float16* ga1 = &A [(size_t)(m0 + rr0 + 64) * K + c8];
    const _Float16* gb0 = &BT[(size_t)(n0 + rr0     ) * K + c8];
    const _Float16* gb1 = &BT[(size_t)(n0 + rr0 + 64) * K + c8];

    unsigned la0[2], la1[2], lb0[2], lb1[2];
#pragma unroll
    for (int b = 0; b < 2; ++b) {
        la0[b] = (unsigned)(size_t)&As[b][rr0     ][c8];
        la1[b] = (unsigned)(size_t)&As[b][rr0 + 64][c8];
        lb0[b] = (unsigned)(size_t)&Bs[b][rr0     ][c8];
        lb1[b] = (unsigned)(size_t)&Bs[b][rr0 + 64][c8];
    }

    // Prologue: async-fill buffer 0 with tile k=0
    async_ld128(la0[0], ga0);
    async_ld128(la1[0], ga1);
    async_ld128(lb0[0], gb0);
    async_ld128(lb1[0], gb1);
    wait_async0();
    __syncthreads();

    int buf = 0;
    for (int k0 = 0; k0 < K; k0 += BK) {
        const int kn = k0 + BK;
        if (kn < K) {
            // async-fill the other buffer with the next tile (overlaps WMMAs)
            async_ld128(la0[buf ^ 1], ga0 + kn);
            async_ld128(la1[buf ^ 1], ga1 + kn);
            async_ld128(lb0[buf ^ 1], gb0 + kn);
            async_ld128(lb1[buf ^ 1], gb1 + kn);
        }

        // Fragments: ISA 16-bit layout => halves [8*hi .. 8*hi+7] and
        // [16+8*hi .. 23+8*hi] -> two b128 LDS loads per fragment.
        Frag fa[2], fb[4];
#pragma unroll
        for (int mi = 0; mi < 2; ++mi) {
            int mrow = waveM * 32 + mi * 16 + nl;
            fa[mi].q[0] = *(const v4u*)&As[buf][mrow][8 * hi];
            fa[mi].q[1] = *(const v4u*)&As[buf][mrow][16 + 8 * hi];
        }
#pragma unroll
        for (int nj = 0; nj < 4; ++nj) {
            int nrow = waveN * 64 + nj * 16 + nl;
            fb[nj].q[0] = *(const v4u*)&Bs[buf][nrow][8 * hi];
            fb[nj].q[1] = *(const v4u*)&Bs[buf][nrow][16 + 8 * hi];
        }

#pragma unroll
        for (int mi = 0; mi < 2; ++mi)
#pragma unroll
            for (int nj = 0; nj < 4; ++nj)
                acc[mi][nj] = __builtin_amdgcn_wmma_f32_16x16x32_f16(
                    false, fa[mi].v, false, fb[nj].v,
                    (short)0, acc[mi][nj], false, false);

        wait_async0();       // next tile resident in other buffer
        __syncthreads();     // all waves done reading current buffer
        buf ^= 1;
    }

    // Store: C layout — vgpr r is row (r + 8*hi), col = nl. Guard-free.
#pragma unroll
    for (int mi = 0; mi < 2; ++mi)
#pragma unroll
        for (int nj = 0; nj < 4; ++nj)
#pragma unroll
            for (int r = 0; r < 8; ++r) {
                int gm = m0 + waveM * 32 + mi * 16 + r + hi * 8;
                int gn = n0 + waveN * 64 + nj * 16 + nl;
                float v = acc[mi][nj][r];
                if (bias)     v += bias[gn];
                if (residual) v += residual[(size_t)gm * N + gn];
                C[(size_t)gm * N + gn] = (OUT_T)v;
            }
}

// ---------------------------------------------------------------------------
// Elementwise / reduction kernels
// ---------------------------------------------------------------------------
__global__ void f2h_kernel(const float* __restrict__ x, _Float16* __restrict__ y, long n) {
    long i = (long)blockIdx.x * 256 + threadIdx.x;
    if (i < n) y[i] = (_Float16)x[i];
}

// Convert weight (K,N) f32 row-major -> BT (Npad,Kpad) f16 row-major, zero-padded.
__global__ void f2h_t_kernel(const float* __restrict__ x, _Float16* __restrict__ y,
                             int K, int N, int Kpad, int Npad) {
    long idx = (long)blockIdx.x * 256 + threadIdx.x;
    if (idx >= (long)Npad * Kpad) return;
    int k = (int)(idx % Kpad); int n = (int)(idx / Kpad);
    float v = (k < K && n < N) ? x[(size_t)k * N + n] : 0.f;
    y[idx] = (_Float16)v;
}

// LayerNorm over rows of length n; one block (256 thr) per row. b may be null.
template <typename OUT_T>
__global__ void ln_kernel(const float* __restrict__ x, const float* __restrict__ g,
                          const float* __restrict__ b, OUT_T* __restrict__ y, int n) {
    const int row = blockIdx.x;
    const float* xr = x + (size_t)row * n;
    __shared__ float s1[256], s2[256];
    float a = 0.f, sq = 0.f;
    for (int i = threadIdx.x; i < n; i += 256) { float v = xr[i]; a += v; sq += v * v; }
    s1[threadIdx.x] = a; s2[threadIdx.x] = sq;
    __syncthreads();
    for (int off = 128; off > 0; off >>= 1) {
        if (threadIdx.x < off) { s1[threadIdx.x] += s1[threadIdx.x + off]; s2[threadIdx.x] += s2[threadIdx.x + off]; }
        __syncthreads();
    }
    float mean = s1[0] / n;
    float var  = s2[0] / n - mean * mean;
    float inv  = rsqrtf(var + 1e-5f);
    OUT_T* yr = y + (size_t)row * n;
    for (int i = threadIdx.x; i < n; i += 256) {
        float v = (xr[i] - mean) * inv * g[i] + (b ? b[i] : 0.f);
        yr[i] = (OUT_T)v;
    }
}

// L2-normalize rows of length n -> f16 (same layout)
__global__ void l2n_rows_kernel(const float* __restrict__ x, _Float16* __restrict__ y, int n) {
    const int row = blockIdx.x;
    const float* xr = x + (size_t)row * n;
    __shared__ float s1[256];
    float sq = 0.f;
    for (int i = threadIdx.x; i < n; i += 256) { float v = xr[i]; sq += v * v; }
    s1[threadIdx.x] = sq; __syncthreads();
    for (int off = 128; off > 0; off >>= 1) {
        if (threadIdx.x < off) s1[threadIdx.x] += s1[threadIdx.x + off];
        __syncthreads();
    }
    float inv = 1.f / fmaxf(sqrtf(s1[0]), 1e-12f);
    for (int i = threadIdx.x; i < n; i += 256)
        y[(size_t)row * n + i] = (_Float16)(xr[i] * inv);
}

// Patchify first frame: out (1024 rows x 256)
__global__ void patch_first_kernel(const float* __restrict__ video, float* __restrict__ out) {
    long idx = (long)blockIdx.x * 256 + threadIdx.x;
    if (idx >= (long)1024 * 256) return;
    int e = idx & 255; int r = (int)(idx >> 8);
    int b = r >> 8, p = r & 255;
    int hh = p >> 4, ww = p & 15;
    int ph = e >> 4, pw = e & 15;
    size_t voff = (size_t)b * 33 * 65536 + (size_t)(hh * 16 + ph) * 256 + (ww * 16 + pw);
    out[idx] = video[voff];
}

// Patchify rest: out (16384 rows x 512); elem e = pt*256 + ph*16 + pw
__global__ void patch_rest_kernel(const float* __restrict__ video, float* __restrict__ out) {
    long idx = (long)blockIdx.x * 256 + threadIdx.x;
    if (idx >= (long)16384 * 512) return;
    int e = idx & 511; int r = (int)(idx >> 9);
    int b = r >> 12, tt = (r >> 8) & 15, p = r & 255;
    int pt = e >> 8, ph = (e >> 4) & 15, pw = e & 15;
    int hh = p >> 4, ww = p & 15;
    int frame = 1 + tt * 2 + pt;
    size_t voff = (size_t)b * 33 * 65536 + (size_t)frame * 65536 +
                  (size_t)(hh * 16 + ph) * 256 + (ww * 16 + pw);
    out[idx] = video[voff];
}

__global__ void scatter_first_kernel(const float* __restrict__ src, float* __restrict__ tokens) {
    long idx = (long)blockIdx.x * 256 + threadIdx.x;
    if (idx >= (long)1024 * 512) return;
    int d = idx & 511; int r = (int)(idx >> 9);
    int b = r >> 8, p = r & 255;
    tokens[((size_t)(b * 4352 + p)) * 512 + d] = src[idx];
}

__global__ void scatter_rest_kernel(const float* __restrict__ src, float* __restrict__ tokens) {
    long idx = (long)blockIdx.x * 256 + threadIdx.x;
    if (idx >= (long)16384 * 512) return;
    int d = idx & 511; int r = (int)(idx >> 9);
    int b = r >> 12, tt = (r >> 8) & 15, p = r & 255;
    tokens[((size_t)(b * 4352 + (1 + tt) * 256 + p)) * 512 + d] = src[idx];
}

// PEG: depthwise causal conv3d + bias + residual; flat buffer viewed as (4,17,16,16,512)
__global__ void peg_kernel(const float* __restrict__ x, const float* __restrict__ w,
                           const float* __restrict__ bconv, float* __restrict__ y) {
    long idx = (long)blockIdx.x * 256 + threadIdx.x;
    if (idx >= (long)ROWS * 512) return;
    int d = idx & 511; int r = (int)(idx >> 9);
    int b_ = r / 4352; int rem = r % 4352;
    int t_ = rem >> 8; int hw = rem & 255;
    int h_ = hw >> 4, w_ = hw & 15;
    float s = bconv[d];
#pragma unroll
    for (int dt = 0; dt < 3; ++dt) {
        int ts = t_ + dt - 2;
        if (ts < 0) continue;
#pragma unroll
        for (int dh = 0; dh < 3; ++dh) {
            int hs = h_ + dh - 1;
            if (hs < 0 || hs > 15) continue;
#pragma unroll
            for (int dw = 0; dw < 3; ++dw) {
                int ws_ = w_ + dw - 1;
                if (ws_ < 0 || ws_ > 15) continue;
                s += x[((size_t)(b_ * 4352 + ts * 256 + hs * 16 + ws_)) * 512 + d] *
                     w[(size_t)d * 27 + dt * 9 + dh * 3 + dw];
            }
        }
    }
    y[idx] = s + x[idx];
}

// Per-(row, head) l2norm + learned per-dim scale for q and k (k = first half of kv row)
__global__ void qknorm_kernel(float* __restrict__ q, float* __restrict__ kv,
                              const float* __restrict__ qs, const float* __restrict__ ks,
                              int rows) {
    int idx = blockIdx.x * 256 + threadIdx.x;
    int hd = idx & 7; int r = idx >> 3;
    if (r >= rows) return;
    float* qp = q + (size_t)r * 512 + hd * 64;
    float* kp = kv + (size_t)r * 1024 + hd * 64;
    float sq = 0.f, sk = 0.f;
#pragma unroll
    for (int d = 0; d < 64; ++d) { sq += qp[d] * qp[d]; sk += kp[d] * kp[d]; }
    float iq = 1.f / fmaxf(sqrtf(sq), 1e-12f);
    float ik = 1.f / fmaxf(sqrtf(sk), 1e-12f);
#pragma unroll
    for (int d = 0; d < 64; ++d) {
        qp[d] = qp[d] * iq * qs[d];
        kp[d] = kp[d] * ik * ks[d];
    }
}

// Flash attention: one block per (seq, head), one thread per query row.
// bias: (8, n, n) or null; causal adds -|i-j| * 2^-(head+1) and masks j > i.
__global__ __launch_bounds__(256) void attn_kernel(
    const float* __restrict__ q, const float* __restrict__ kv, float* __restrict__ out,
    int nseq, int n, const float* __restrict__ bias, int causal)
{
    int sh = blockIdx.x;
    int head = sh & 7, seq = sh >> 3;
    int i = threadIdx.x;
    if (i >= n) return;
    const float* qp = q + (size_t)(seq * n + i) * 512 + head * 64;
    float qr[64], acc[64];
#pragma unroll
    for (int d = 0; d < 64; ++d) { qr[d] = qp[d]; acc[d] = 0.f; }
    float m = -3.4e38f, l = 0.f;
    float slope = exp2f(-(float)(head + 1));
    int jmax = causal ? i : (n - 1);
    for (int j = 0; j <= jmax; ++j) {
        const float* kp = kv + (size_t)(seq * n + j) * 1024 + head * 64;
        float s = 0.f;
#pragma unroll
        for (int d = 0; d < 64; ++d) s += qr[d] * kp[d];
        s *= 8.0f;
        if (bias) s += bias[((size_t)head * n + i) * n + j];
        if (causal) s -= fabsf((float)(i - j)) * slope;
        float mn = fmaxf(m, s);
        float sc = expf(m - mn);
        float p  = expf(s - mn);
        l = l * sc + p;
        const float* vp = kp + 512;
#pragma unroll
        for (int d = 0; d < 64; ++d) acc[d] = acc[d] * sc + p * vp[d];
        m = mn;
    }
    float inv = 1.f / l;
    float* op = out + (size_t)(seq * n + i) * 512 + head * 64;
#pragma unroll
    for (int d = 0; d < 64; ++d) op[d] = acc[d] * inv;
}

// gated = a * gelu(gate); hidden (rows x FFH_PAD f16, real split at 1365)
// -> gated (rows x FFI_PAD f16, zero-padded cols [1365,1376))
__global__ void gelugate_kernel(const _Float16* __restrict__ hidden,
                                _Float16* __restrict__ gated, int rows) {
    long idx = (long)blockIdx.x * 256 + threadIdx.x;
    if (idx >= (long)rows * FFI_PAD) return;
    int c = (int)(idx % FFI_PAD); int r = (int)(idx / FFI_PAD);
    float o = 0.f;
    if (c < FFI_) {
        float a = (float)hidden[(size_t)r * FFH_PAD + c];
        float g = (float)hidden[(size_t)r * FFH_PAD + FFI_ + c];
        o = a * 0.5f * g * (1.f + erff(g * 0.70710678118f));
    }
    gated[(size_t)r * FFI_PAD + c] = (_Float16)o;
}

// CPB layer0: rel features -> leaky_relu(rel @ w0 + b0), f16 out, chunked over rows
__global__ void cpb_l0_kernel(const float* __restrict__ w0, const float* __restrict__ b0,
                              _Float16* __restrict__ feat, int base, int rows) {
    long idx = (long)blockIdx.x * 256 + threadIdx.x;
    if (idx >= (long)rows * 512) return;
    int d = idx & 511; int rr = (int)(idx >> 9);
    int r = base + rr;
    int p = r >> 8, qq = r & 255;
    float r0 = (float)((p >> 4) - (qq >> 4));
    float r1 = (float)((p & 15) - (qq & 15));
    float f0 = (r0 > 0.f) ? log1pf(r0) : ((r0 < 0.f) ? -log1pf(-r0) : 0.f);
    float f1 = (r1 > 0.f) ? log1pf(r1) : ((r1 < 0.f) ? -log1pf(-r1) : 0.f);
    float v = f0 * w0[d] + f1 * w0[512 + d] + b0[d];
    feat[idx] = (_Float16)(v > 0.f ? v : 0.1f * v);
}

__global__ void lrelu_f2h_kernel(const float* __restrict__ x, _Float16* __restrict__ y, long n) {
    long i = (long)blockIdx.x * 256 + threadIdx.x;
    if (i < n) { float v = x[i]; y[i] = (_Float16)(v > 0.f ? v : 0.1f * v); }
}

// CPB final: (rows x 512 f16) @ w2 (512 x 8) + b2 -> bias[hd*65536 + row]
__global__ void cpb_l2_kernel(const _Float16* __restrict__ act, const float* __restrict__ w2,
                              const float* __restrict__ b2, float* __restrict__ bias_out,
                              int base, int rows) {
    int idx = blockIdx.x * 256 + threadIdx.x;
    int hd = idx & 7; int rr = idx >> 3;
    if (rr >= rows) return;
    const _Float16* ar = act + (size_t)rr * 512;
    float s = b2[hd];
    for (int k = 0; k < 512; ++k) s += (float)ar[k] * w2[k * 8 + hd];
    bias_out[(size_t)hd * 65536 + base + rr] = s;
}

// (b,t,hw,D) -> (b,hw,t,D)
__global__ void t_fwd_kernel(const float* __restrict__ src, float* __restrict__ dst) {
    long idx = (long)blockIdx.x * 256 + threadIdx.x;
    if (idx >= (long)ROWS * 512) return;
    int d = idx & 511; int r = (int)(idx >> 9);
    int t = r % 17; int bhw = r / 17;
    int hw = bhw & 255; int b = bhw >> 8;
    dst[idx] = src[(((size_t)(b * 17 + t)) * 256 + hw) * 512 + d];
}

// (b,hw,t,D) -> (b,t,hw,D)
__global__ void t_bwd_kernel(const float* __restrict__ src, float* __restrict__ dst) {
    long idx = (long)blockIdx.x * 256 + threadIdx.x;
    if (idx >= (long)ROWS * 512) return;
    int d = idx & 511; int r = (int)(idx >> 9);
    int hw = r & 255; int bt = r >> 8;
    int t = bt % 17; int b = bt / 17;
    dst[idx] = src[(((size_t)(b * 256 + hw)) * 17 + t) * 512 + d];
}

// Argmax over n scores per row
__global__ void argmax_kernel(const float* __restrict__ scores, int* __restrict__ ind, int n) {
    const int row = blockIdx.x;
    const float* sr = scores + (size_t)row * n;
    __shared__ float bv[256]; __shared__ int bi[256];
    float best = -3.4e38f; int bidx = 0;
    for (int i = threadIdx.x; i < n; i += 256) {
        float v = sr[i];
        if (v > best) { best = v; bidx = i; }
    }
    bv[threadIdx.x] = best; bi[threadIdx.x] = bidx;
    __syncthreads();
    for (int off = 128; off > 0; off >>= 1) {
        if (threadIdx.x < off) {
            if (bv[threadIdx.x + off] > bv[threadIdx.x]) {
                bv[threadIdx.x] = bv[threadIdx.x + off];
                bi[threadIdx.x] = bi[threadIdx.x + off];
            }
        }
        __syncthreads();
    }
    if (threadIdx.x == 0) ind[row] = bi[0];
}

// out[row] = l2norm(codebook[ind[row]])  (straight-through forward value)
__global__ void gather_kernel(const int* __restrict__ ind, const float* __restrict__ cb,
                              float* __restrict__ out) {
    const int row = blockIdx.x;
    const float* cr = cb + (size_t)ind[row] * 512;
    __shared__ float s1[256];
    float sq = 0.f;
    for (int i = threadIdx.x; i < 512; i += 256) { float v = cr[i]; sq += v * v; }
    s1[threadIdx.x] = sq; __syncthreads();
    for (int off = 128; off > 0; off >>= 1) {
        if (threadIdx.x < off) s1[threadIdx.x] += s1[threadIdx.x + off];
        __syncthreads();
    }
    float inv = 1.f / fmaxf(sqrtf(s1[0]), 1e-12f);
    for (int i = threadIdx.x; i < 512; i += 256)
        out[(size_t)row * 512 + i] = cr[i] * inv;
}

// ---------------------------------------------------------------------------
// Host orchestration
// ---------------------------------------------------------------------------
extern "C" void kernel_launch(void* const* d_in, const int* in_sizes, int n_in,
                              void* d_out, int out_size, void* d_ws, size_t ws_size,
                              hipStream_t stream) {
    (void)in_sizes; (void)n_in; (void)out_size; (void)ws_size;

    const float* video      = (const float*)d_in[0];
    const float* pef_ln_g   = (const float*)d_in[1];
    const float* pef_ln_b   = (const float*)d_in[2];
    const float* pef_w      = (const float*)d_in[3];
    const float* pef_bias   = (const float*)d_in[4];
    const float* pef_ln2_g  = (const float*)d_in[5];
    const float* pef_ln2_b  = (const float*)d_in[6];
    const float* per_ln_g   = (const float*)d_in[7];
    const float* per_ln_b   = (const float*)d_in[8];
    const float* per_w      = (const float*)d_in[9];
    const float* per_bias   = (const float*)d_in[10];
    const float* per_ln2_g  = (const float*)d_in[11];
    const float* per_ln2_b  = (const float*)d_in[12];
    const float* cpb_w0     = (const float*)d_in[13];
    const float* cpb_b0     = (const float*)d_in[14];
    const float* cpb_w1     = (const float*)d_in[15];
    const float* cpb_b1     = (const float*)d_in[16];
    const float* cpb_w2     = (const float*)d_in[17];
    const float* cpb_b2     = (const float*)d_in[18];
    const float* codebook   = (const float*)d_in[19];
    const float* sp[13]; for (int i = 0; i < 13; ++i) sp[i] = (const float*)d_in[20 + i];
    const float* tp[13]; for (int i = 0; i < 13; ++i) tp[i] = (const float*)d_in[33 + i];
    // P[]: 0 peg_w 1 peg_b 2 attn_g 3 q_w 4 kv_w 5 qs 6 ks 7 out_w 8 ff_g 9 ff_b 10 ff_w1 11 ff_w2 12 norm_g

    // workspace allocator
    char* wbase = (char*)d_ws;
    size_t wcur = 0;
    auto alloc = [&](size_t bytes) -> void* {
        wcur = (wcur + 255) & ~(size_t)255;
        void* p = wbase + wcur; wcur += bytes; return p;
    };

    float*     bufA   = (float*)alloc((size_t)ROWS * 512 * 4);
    float*     bufB   = (float*)alloc((size_t)ROWS * 512 * 4);
    _Float16*  xn16   = (_Float16*)alloc((size_t)ROWS * 512 * 2);
    float*     qbuf   = (float*)alloc((size_t)ROWS * 512 * 4);
    float*     kvbuf  = (float*)alloc((size_t)ROWS * 1024 * 4);       // also VQ scores (2176x8192)
    float*     aout   = (float*)alloc((size_t)ROWS * 512 * 4);        // attn out / patch rows / cpb y1
    _Float16*  hid16  = (_Float16*)alloc((size_t)CHUNK * FFH_PAD * 2);// ffn hidden / embed ln16 / cpb feat-act
    _Float16*  gat16  = (_Float16*)alloc((size_t)CHUNK * FFI_PAD * 2);
    float*     cpbb   = (float*)alloc((size_t)8 * 65536 * 4);
    _Float16*  wq16   = (_Float16*)alloc((size_t)512 * 512 * 2);      // BT layouts
    _Float16*  wkv16  = (_Float16*)alloc((size_t)1024 * 512 * 2);
    _Float16*  wo16   = (_Float16*)alloc((size_t)512 * 512 * 2);
    _Float16*  wf116  = (_Float16*)alloc((size_t)FFH_PAD * 512 * 2);
    _Float16*  wf216  = (_Float16*)alloc((size_t)512 * FFI_PAD * 2);
    _Float16*  pw16   = (_Float16*)alloc((size_t)512 * 512 * 2);      // pef/per/cpb_w1 (sequential reuse)
    _Float16*  cbn16  = (_Float16*)alloc((size_t)CBN * 512 * 2);      // l2-normalized codebook (natural BT)
    int*       ind    = (int*)alloc((size_t)ROWS * 4);

    auto EW = [](long n) { return dim3((unsigned)((n + 255) / 256)); };
    auto tohalfT = [&](const float* x, _Float16* y, int K, int N, int Kpad, int Npad) {
        f2h_t_kernel<<<EW((long)Npad * Kpad), 256, 0, stream>>>(x, y, K, N, Kpad, Npad);
    };
    auto gemm32 = [&](const _Float16* A, const _Float16* BT, float* C, int M, int N, int K,
                      const float* bias, const float* res) {
        dim3 g(N / BN, M / BM);
        gemm_kernel<float><<<g, 256, 0, stream>>>(A, BT, C, M, N, K, bias, res);
    };
    auto gemm16 = [&](const _Float16* A, const _Float16* BT, _Float16* C, int M, int N, int K) {
        dim3 g(N / BN, M / BM);
        gemm_kernel<_Float16><<<g, 256, 0, stream>>>(A, BT, C, M, N, K, nullptr, nullptr);
    };

    // ---- Patch embed ----
    patch_first_kernel<<<EW((long)1024 * 256), 256, 0, stream>>>(video, aout);
    ln_kernel<_Float16><<<1024, 256, 0, stream>>>(aout, pef_ln_g, pef_ln_b, hid16, 256);
    tohalfT(pef_w, pw16, 256, 512, 256, 512);
    gemm32(hid16, pw16, qbuf, 1024, 512, 256, pef_bias, nullptr);
    ln_kernel<float><<<1024, 256, 0, stream>>>(qbuf, pef_ln2_g, pef_ln2_b, aout, 512);
    scatter_first_kernel<<<EW((long)1024 * 512), 256, 0, stream>>>(aout, bufA);

    patch_rest_kernel<<<EW((long)16384 * 512), 256, 0, stream>>>(video, aout);
    ln_kernel<_Float16><<<16384, 256, 0, stream>>>(aout, per_ln_g, per_ln_b, hid16, 512);
    tohalfT(per_w, pw16, 512, 512, 512, 512);
    gemm32(hid16, pw16, qbuf, 16384, 512, 512, per_bias, nullptr);
    ln_kernel<float><<<16384, 256, 0, stream>>>(qbuf, per_ln2_g, per_ln2_b, aout, 512);
    scatter_rest_kernel<<<EW((long)16384 * 512), 256, 0, stream>>>(aout, bufA);

    // ---- CPB relative-position bias (8,256,256), chunked 4 x 16384 rows ----
    tohalfT(cpb_w1, pw16, 512, 512, 512, 512);
    for (int c = 0; c < 4; ++c) {
        int base = c * 16384;
        cpb_l0_kernel<<<EW((long)16384 * 512), 256, 0, stream>>>(cpb_w0, cpb_b0, hid16, base, 16384);
        gemm32(hid16, pw16, aout, 16384, 512, 512, cpb_b1, nullptr);
        lrelu_f2h_kernel<<<EW((long)16384 * 512), 256, 0, stream>>>(aout, hid16, (long)16384 * 512);
        cpb_l2_kernel<<<EW((long)16384 * 8), 256, 0, stream>>>(hid16, cpb_w2, cpb_b2, cpbb, base, 16384);
    }

    // ---- Transformer stack (shared between spatial / temporal) ----
    float* cur = bufA; float* alt = bufB;
    auto run_stack = [&](const float** P, const float* bias, int causal, int nseq, int nctx) {
        for (int i = 0; i < 4; ++i) {
            tohalfT(P[3]  + (size_t)i * 512 * 512,   wq16,  512, 512, 512, 512);
            tohalfT(P[4]  + (size_t)i * 512 * 1024,  wkv16, 512, 1024, 512, 1024);
            tohalfT(P[7]  + (size_t)i * 512 * 512,   wo16,  512, 512, 512, 512);
            tohalfT(P[10] + (size_t)i * 512 * FFH,   wf116, 512, FFH, 512, FFH_PAD);
            tohalfT(P[11] + (size_t)i * FFI_ * 512,  wf216, FFI_, 512, FFI_PAD, 512);

            // PEG + residual (writes alt), then swap
            peg_kernel<<<EW((long)ROWS * 512), 256, 0, stream>>>(
                cur, P[0] + (size_t)i * 512 * 27, P[1] + (size_t)i * 512, alt);
            { float* t = cur; cur = alt; alt = t; }

            // Attention
            ln_kernel<_Float16><<<ROWS, 256, 0, stream>>>(cur, P[2] + (size_t)i * 512, nullptr, xn16, 512);
            gemm32(xn16, wq16,  qbuf,  ROWS, 512,  512, nullptr, nullptr);
            gemm32(xn16, wkv16, kvbuf, ROWS, 1024, 512, nullptr, nullptr);
            qknorm_kernel<<<EW((long)ROWS * 8), 256, 0, stream>>>(
                qbuf, kvbuf, P[5] + (size_t)i * 64, P[6] + (size_t)i * 64, ROWS);
            attn_kernel<<<nseq * HEADS, 256, 0, stream>>>(qbuf, kvbuf, aout, nseq, nctx, bias, causal);
            f2h_kernel<<<EW((long)ROWS * 512), 256, 0, stream>>>(aout, xn16, (long)ROWS * 512);
            gemm32(xn16, wo16, cur, ROWS, 512, 512, nullptr, cur);   // + residual, in place

            // FFN (chunked over rows)
            ln_kernel<_Float16><<<ROWS, 256, 0, stream>>>(cur, P[8] + (size_t)i * 512, P[9] + (size_t)i * 512, xn16, 512);
            for (int ch = 0; ch < 4; ++ch) {
                size_t ro = (size_t)ch * CHUNK;
                gemm16(xn16 + ro * 512, wf116, hid16, CHUNK, FFH_PAD, 512);
                gelugate_kernel<<<EW((long)CHUNK * FFI_PAD), 256, 0, stream>>>(hid16, gat16, CHUNK);
                gemm32(gat16, wf216, cur + ro * 512, CHUNK, 512, FFI_PAD, nullptr, cur + ro * 512);
            }
        }
    };

    // Spatial stack (68 seqs x 256 tokens, CPB bias)
    run_stack(sp, cpbb, 0, 68, 256);
    ln_kernel<float><<<ROWS, 256, 0, stream>>>(cur, sp[12], nullptr, alt, 512);   // cur==bufA, alt==bufB
    t_fwd_kernel<<<EW((long)ROWS * 512), 256, 0, stream>>>(alt, cur);             // -> (b,hw,t,D) in bufA

    // Temporal stack (1024 seqs x 17 tokens, causal + alibi)
    run_stack(tp, nullptr, 1, 1024, 17);
    ln_kernel<float><<<ROWS, 256, 0, stream>>>(cur, tp[12], nullptr, alt, 512);
    t_bwd_kernel<<<EW((long)ROWS * 512), 256, 0, stream>>>(alt, cur);             // -> (b,t,hw,D) in bufA

    // ---- Cosine-sim VQ ----
    l2n_rows_kernel<<<ROWS, 256, 0, stream>>>(cur, xn16, 512);
    l2n_rows_kernel<<<CBN, 256, 0, stream>>>(codebook, cbn16, 512);               // natural (N,K) layout
    for (int c = 0; c < 8; ++c) {
        int base = c * SCORE_CHUNK;
        gemm32(xn16 + (size_t)base * 512, cbn16, kvbuf, SCORE_CHUNK, CBN, 512, nullptr, nullptr);
        argmax_kernel<<<SCORE_CHUNK, 256, 0, stream>>>(kvbuf, ind + base, CBN);
    }
    gather_kernel<<<ROWS, 256, 0, stream>>>(ind, codebook, (float*)d_out);
}
